// QuantumSelfAttention_65481071408154
// MI455X (gfx1250) — matrix operand order repaired and checked by reading, not compiled, not run on hardware
//
#include <hip/hip_runtime.h>
#include <hip/hip_bf16.h>

// ---------------------------------------------------------------------------
// QuantumSelfAttention, B=8 L=4096 D=1024 H=16 DK=64
//   m[l,d]  = mean_{b,h} x[b,l,h*64+d]                      [4096,64]
//   attn    = softmax(sin^2(clip(m m^T, -pi, pi)/2))        [4096,4096] shared
//   out[b]  = attn @ x[b]                                   (heads collapse!)
// Dominant cost: 8x (4096x1024x4096) GEMM = 275 GFLOP -> bf16 WMMA w/ f32 acc,
// B-tile staged in LDS via gfx1250 async global->LDS DMA (double buffered).
// Block tile 256(M)x128(N) keeps total L2 traffic ~3 GiB (attn+xt live in the
// 192 MB L2), balancing the WMMA compute time.
// m m^T (2.1 GFLOP, accuracy-sensitive) -> f32 WMMA 16x16x4.
// ---------------------------------------------------------------------------

#define Lq 4096
#define Dq 1024
#define Bq 8
#define DKq 64

typedef __attribute__((ext_vector_type(2)))  float  v2f;
typedef __attribute__((ext_vector_type(8)))  float  v8f;
typedef __attribute__((ext_vector_type(8)))  __bf16 v8bf;
typedef __attribute__((ext_vector_type(16))) __bf16 v16bf;
typedef int v4i __attribute__((vector_size(16)));

#define PI_F 3.14159265358979323846f

// ---- gfx1250 async global->LDS support (guarded; sync fallback) -----------
#if defined(__has_builtin)
#if __has_builtin(__builtin_amdgcn_global_load_async_to_lds_b128)
#define QSA_ASYNC 1
#endif
#if __has_builtin(__builtin_amdgcn_s_wait_asynccnt)
#define QSA_WAITB 1
#endif
#endif

__device__ __forceinline__ void qsa_copy_b128(const __bf16* g, __bf16* l) {
#ifdef QSA_ASYNC
    // Param1 is AS(1) v4i* (per hipcc diagnostic). Generic & global share the
    // numeric address -> integer round-trip also drops const legally.
    __attribute__((address_space(1))) v4i* gp =
        (__attribute__((address_space(1))) v4i*)(uintptr_t)g;
    // LDS must be a true addrspacecast (generic LDS ptrs carry the aperture).
    __attribute__((address_space(3))) v4i* lp =
        (__attribute__((address_space(3))) v4i*)l;
    __builtin_amdgcn_global_load_async_to_lds_b128(gp, lp, 0, 0);
#else
    *(v8bf*)l = *(const v8bf*)g;
#endif
}

__device__ __forceinline__ void qsa_wait_async() {
#ifdef QSA_ASYNC
#ifdef QSA_WAITB
    __builtin_amdgcn_s_wait_asynccnt(0);
#else
    asm volatile("s_wait_asynccnt 0x0" ::: "memory");
#endif
#endif
}

// ---------------------------------------------------------------------------
// Kernel 1: m[l,d] = (1/128) sum_{b,h} x[b,l,h*64+d].  grid=4096, block=64
// ---------------------------------------------------------------------------
__global__ void qsa_mean(const float* __restrict__ x, float* __restrict__ m) {
    const int l = blockIdx.x;
    const int d = threadIdx.x;              // 0..63
    float s = 0.0f;
    for (int b = 0; b < Bq; ++b) {
        const float* xr = x + ((size_t)b * Lq + l) * Dq;
#pragma unroll
        for (int h = 0; h < 16; ++h) s += xr[h * DKq + d];
    }
    m[l * DKq + d] = s * (1.0f / 128.0f);
}

// ---------------------------------------------------------------------------
// Kernel 2a: E[i,j] = exp(sin^2(clip(m[i,:]·m[j,:])/2)).  f32 WMMA 16x16x4.
// grid=(32,256), block=256 (8 waves; wave w -> j-tile blockIdx.x*8+w,
// i-tile blockIdx.y). Per tile: 16 K-steps of v_wmma_f32_16x16x4_f32.
// A frag (16x4): lane p holds M=p%16, K=2*(p/16)+{0,1}; B mirrors with N=p%16.
// ---------------------------------------------------------------------------
__global__ void qsa_logits_exp(const float* __restrict__ m,
                               float* __restrict__ E) {
    const int lane = threadIdx.x & 31;
    const int wave = threadIdx.x >> 5;
    const int j0 = (blockIdx.x * 8 + wave) * 16;
    const int i0 = blockIdx.y * 16;
    const int lm = lane & 15;
    const int kb = 2 * (lane >> 4);

    const float* arow = m + (size_t)(i0 + lm) * DKq + kb;
    const float* brow = m + (size_t)(j0 + lm) * DKq + kb;

    v8f acc = {};
#pragma unroll
    for (int kk = 0; kk < DKq; kk += 4) {
        v2f a, b;
        a.x = arow[kk]; a.y = arow[kk + 1];
        b.x = brow[kk]; b.y = brow[kk + 1];
        acc = __builtin_amdgcn_wmma_f32_16x16x4_f32(
            false, a, false, b, (short)0, acc, false, false);
    }

    // C layout: VGPR r -> M = i0 + r + 8*(lane/16), N = j0 + lane%16
    const int nj = j0 + lm;
#pragma unroll
    for (int r = 0; r < 8; ++r) {
        const int ii = i0 + r + 8 * (lane >> 4);
        float g = acc[r];
        g = fminf(fmaxf(g, -PI_F), PI_F);
        const float s = __sinf(0.5f * g);
        // logits in [0,1] -> exp without max-subtraction is safe
        E[(size_t)ii * Lq + nj] = __expf(s * s);
    }
}

// ---------------------------------------------------------------------------
// Kernel 2b: attn[row,:] = E[row,:] / sum(E[row,:]) as bf16.
// One block (256 threads) per row; deterministic tree reduction (no atomics).
// ---------------------------------------------------------------------------
__global__ void qsa_softmax_norm(const float* __restrict__ E,
                                 __bf16* __restrict__ A) {
    const int row = blockIdx.x;
    const float* e = E + (size_t)row * Lq;
    __shared__ float red[256];
    float s = 0.0f;
    for (int j = threadIdx.x; j < Lq; j += 256) s += e[j];
    red[threadIdx.x] = s;
    __syncthreads();
    for (int off = 128; off > 0; off >>= 1) {
        if (threadIdx.x < off) red[threadIdx.x] += red[threadIdx.x + off];
        __syncthreads();
    }
    const float inv = 1.0f / red[0];
    __bf16* a = A + (size_t)row * Lq;
    for (int j = threadIdx.x; j < Lq; j += 256)
        a[j] = (__bf16)(e[j] * inv);
}

// ---------------------------------------------------------------------------
// Kernel 2c: xt[b][d][l] = bf16(x[b][l][d])  (transpose so the GEMM's B
// fragments become contiguous 16B loads). 32x32 LDS tile transpose.
// grid=(128,32,8), block=(32,8)
// ---------------------------------------------------------------------------
__global__ void qsa_transpose_bf16(const float* __restrict__ x,
                                   __bf16* __restrict__ xt) {
    __shared__ __bf16 tile[32][33];
    const int b = blockIdx.z;
    const int l0 = blockIdx.x * 32;
    const int d0 = blockIdx.y * 32;
    const float* xb = x + (size_t)b * Lq * Dq;
    __bf16* xo = xt + (size_t)b * Dq * Lq;
    const int tx = threadIdx.x;   // 0..31
    const int ty = threadIdx.y;   // 0..7
#pragma unroll
    for (int i = ty; i < 32; i += 8)
        tile[i][tx] = (__bf16)xb[(size_t)(l0 + i) * Dq + d0 + tx];
    __syncthreads();
#pragma unroll
    for (int i = ty; i < 32; i += 8)
        xo[(size_t)(d0 + i) * Lq + l0 + tx] = tile[tx][i];
}

// ---------------------------------------------------------------------------
// Kernel 3: out[b] = attn(bf16) @ x[b](bf16), f32 accumulate.
// v_wmma_f32_16x16x32_bf16. Block = 512 thr = 16 waves; block tile
// 256(M)x128(N); wave tile 16(M)x128(N) = 8 accumulators. K in slabs of 64.
// B tile (128 xt-rows x 64 halves) double-buffered in LDS, filled with
// global_load_async_to_lds_b128 (ASYNCcnt) and read as ds_load_b128 frags.
// Row stride 72 halves (144B): 16B-aligned, bank-conflict-free for 16 rows.
// A frag lane p: row i0+p%16, halves k+(p/16)*8..+7 and k+16+(p/16)*8..+7
// (contiguous 16B global loads, rows wave-exclusive, hot in L2).
// grid=(8,16,8), block=512
// ---------------------------------------------------------------------------
#define KSTEP 64
#define BSTRIDE 72   // halves per LDS row (144 bytes)
#define NTILES 8     // 8 x 16 = 128 N columns per block

__device__ __forceinline__ v16bf qsa_frag(const __bf16* p0, const __bf16* p1) {
    v8bf lo = *(const v8bf*)p0;
    v8bf hi = *(const v8bf*)p1;
    v16bf r;
#pragma unroll
    for (int i = 0; i < 8; ++i) { r[i] = lo[i]; r[i + 8] = hi[i]; }
    return r;
}

__global__ __launch_bounds__(512)
void qsa_gemm(const __bf16* __restrict__ attn,
              const __bf16* __restrict__ xt,
              float* __restrict__ out) {
    __shared__ __bf16 bbuf[2][128 * BSTRIDE];

    const int lane = threadIdx.x & 31;
    const int wave = threadIdx.x >> 5;          // 0..15
    const int b  = blockIdx.z;
    const int i0 = blockIdx.y * 256 + wave * 16;
    const int n0 = blockIdx.x * 128;
    const int lm = lane & 15;
    const int lh = lane >> 4;

    const __bf16* arow = attn + (size_t)(i0 + lm) * Lq;
    const __bf16* xrows = xt + (size_t)b * Dq * Lq + (size_t)n0 * Lq;

    v8f acc[NTILES];
#pragma unroll
    for (int tn = 0; tn < NTILES; ++tn) acc[tn] = v8f{};

    // Each thread moves 2 x 16B chunks per slab: 128 rows x 8 chunks = 1024.
    auto issue = [&](int bs, int kbase) {
        int c = threadIdx.x * 2;
#pragma unroll
        for (int e = 0; e < 2; ++e, ++c) {
            const int row = c >> 3;            // 0..127 (xt row = n0+row)
            const int off = (c & 7) * 8;       // halves 0..56
            qsa_copy_b128(xrows + (size_t)row * Lq + kbase + off,
                          &bbuf[bs][row * BSTRIDE + off]);
        }
    };

    issue(0, 0);

    for (int it = 0; it < Lq / KSTEP; ++it) {
        const int k0 = it * KSTEP;
        qsa_wait_async();          // my slab-(it) DMAs have landed
        __syncthreads();           // everyone's landed; prev compute done
        if (k0 + KSTEP < Lq) issue((it + 1) & 1, k0 + KSTEP);
        __builtin_prefetch(arow + k0 + 2 * KSTEP, 0, 1);   // global_prefetch_b8

        const __bf16* bcur = bbuf[it & 1];
#pragma unroll
        for (int koff = 0; koff < KSTEP; koff += 32) {
            const v16bf afrag = qsa_frag(arow + k0 + koff + lh * 8,
                                         arow + k0 + koff + 16 + lh * 8);
#pragma unroll
            for (int tn = 0; tn < NTILES; ++tn) {
                const __bf16* bb =
                    bcur + (tn * 16 + lm) * BSTRIDE + koff + lh * 8;
                const v16bf bfrag = qsa_frag(bb, bb + 16);
                acc[tn] = __builtin_amdgcn_wmma_f32_16x16x32_bf16(
                    false, afrag, false, bfrag, (short)0, acc[tn],
                    false, false);
            }
        }
        __syncthreads();           // done reading slab before it is re-filled
    }

#pragma unroll
    for (int tn = 0; tn < NTILES; ++tn) {
#pragma unroll
        for (int r = 0; r < 8; ++r) {
            const int ii = i0 + r + 8 * lh;
            const int nn = n0 + tn * 16 + lm;
            out[((size_t)b * Lq + ii) * Dq + nn] = acc[tn][r];
        }
    }
}

// ---------------------------------------------------------------------------
// Workspace layout (bytes):
//   [0,            1 MiB)  : m        (4096*64 f32)
//   [1 MiB,       33 MiB)  : attn     (4096*4096 bf16)
//   [33 MiB,      97 MiB)  : E (f32, 64 MiB)  -- reused as xt (bf16, 64 MiB)
// Total 97 MiB. All kernels on `stream`, fully deterministic.
// ---------------------------------------------------------------------------
extern "C" void kernel_launch(void* const* d_in, const int* in_sizes, int n_in,
                              void* d_out, int out_size, void* d_ws, size_t ws_size,
                              hipStream_t stream) {
    (void)in_sizes; (void)n_in; (void)out_size; (void)ws_size;
    const float* x = (const float*)d_in[0];
    float* out = (float*)d_out;

    char* ws = (char*)d_ws;
    float*  m    = (float*)ws;
    __bf16* attn = (__bf16*)(ws + ((size_t)1 << 20));
    char*   big  = ws + ((size_t)33 << 20);
    float*  E    = (float*)big;
    __bf16* xt   = (__bf16*)big;   // reuses E region after softmax_norm

    hipLaunchKernelGGL(qsa_mean, dim3(Lq), dim3(64), 0, stream, x, m);
    hipLaunchKernelGGL(qsa_logits_exp, dim3(32, 256), dim3(256), 0, stream, m, E);
    hipLaunchKernelGGL(qsa_softmax_norm, dim3(Lq), dim3(256), 0, stream, E, attn);
    hipLaunchKernelGGL(qsa_transpose_bf16, dim3(128, 32, 8), dim3(32, 8), 0, stream, x, xt);
    hipLaunchKernelGGL(qsa_gemm, dim3(8, 16, 8), dim3(512), 0, stream, attn, xt, out);
}
